// MultiHeadAttention_21285857919300
// MI455X (gfx1250) — matrix operand order repaired and checked by reading, not compiled, not run on hardware
//
#include <hip/hip_runtime.h>
#include <hip/hip_bf16.h>

// Problem constants
#define BB   2
#define TT   2048
#define DMM  1024
#define HHN  16
#define DKK  64
#define DVV  64

typedef __attribute__((ext_vector_type(16))) __bf16 v16bf;
typedef __attribute__((ext_vector_type(8)))  float  v8f;

union FragU { v16bf v; float4 q[2]; };

// A-fragment (16x32 bf16): lane holds row m = lane%16.
// elements 0..7  -> K = kbase + 8*half + 0..7
// elements 8..15 -> K = kbase + 16 + 8*half + 0..7
__device__ __forceinline__ v16bf load_a_frag(const __bf16* row_base, int half, int kbase) {
    FragU f;
    const __bf16* p = row_base + kbase + 8 * half;
    f.q[0] = *(const float4*)(p);
    f.q[1] = *(const float4*)(p + 16);
    return f.v;
}

// B-fragment (32x16 bf16): lane holds column n = lane%16.
// elements 0..7  -> K = kbase + 8*half + 0..7
// elements 8..15 -> K = kbase + 16 + 8*half + 0..7
__device__ __forceinline__ v16bf load_b_frag(const __bf16* col_base, int half, int kbase) {
    FragU f;
    const __bf16* p = col_base + kbase + 8 * half;
    f.q[0] = *(const float4*)(p);
    f.q[1] = *(const float4*)(p + 16);
    return f.v;
}

__device__ __forceinline__ v8f wmma_bf16(v16bf a, v16bf b, v8f c) {
    return __builtin_amdgcn_wmma_f32_16x16x32_bf16(
        /*neg_a=*/false, a, /*neg_b=*/false, b,
        /*c_mod=*/(short)0, c, /*reuse_a=*/false, /*reuse_b=*/false);
}

// ---------------------------------------------------------------------------
// Async global -> LDS copy (CDNA5, ASYNCcnt-tracked). One b128 per lane.
// LDS byte offset = low 32 bits of the generic pointer (ISA 10.2 aperture).
// ---------------------------------------------------------------------------
__device__ __forceinline__ void async_b128(const void* g, unsigned lds_off) {
    asm volatile("global_load_async_to_lds_b128 %0, %1, off"
                 :: "v"(lds_off), "v"(g) : "memory");
}
__device__ __forceinline__ void wait_async_0() {
    asm volatile("s_wait_asynccnt 0" ::: "memory");
}
__device__ __forceinline__ void wait_async_16() {
    asm volatile("s_wait_asynccnt 16" ::: "memory");
}

// Stage one 32-key chunk of K (32x64 bf16, contiguous 4KB) and V^T
// (64 dv-rows x 32 keys, strided) into LDS. 16 async b128 ops total.
__device__ __forceinline__ void stage_chunk(const __bf16* Kb, const __bf16* Vtb, int base,
                                            __bf16* kb, __bf16* vb, int lane) {
    const __bf16* gk = Kb + (size_t)base * DKK;
    unsigned lk = (unsigned)(uintptr_t)kb;
#pragma unroll
    for (int i = 0; i < 8; ++i)
        async_b128(gk + (size_t)(i * 32 + lane) * 8, lk + (unsigned)(i * 32 + lane) * 16);
    unsigned lv = (unsigned)(uintptr_t)vb;
#pragma unroll
    for (int i = 0; i < 8; ++i) {
        int idx = i * 32 + lane;
        int row = idx >> 2;          // dv row 0..63
        int c   = idx & 3;           // 16B piece within 64B row
        async_b128(Vtb + (size_t)row * TT + base + c * 8,
                   lv + (unsigned)(row * 64 + c * 16));
    }
}

// ---------------------------------------------------------------------------
// Kernel 1: elementwise f32 -> bf16
// ---------------------------------------------------------------------------
__global__ void cvt_bf16_kernel(const float* __restrict__ in, __bf16* __restrict__ out, int n) {
    int i = blockIdx.x * blockDim.x + threadIdx.x;
    int stride = gridDim.x * blockDim.x;
    for (; i < n; i += stride) out[i] = (__bf16)in[i];
}

// ---------------------------------------------------------------------------
// Kernel 2: W[h][d][k] (f32) -> Wt[h][k][d] (bf16)
// ---------------------------------------------------------------------------
__global__ void cvt_transpose_w_kernel(const float* __restrict__ W, __bf16* __restrict__ Wt,
                                       int D, int K, int n) {
    int i = blockIdx.x * blockDim.x + threadIdx.x;
    int stride = gridDim.x * blockDim.x;
    for (; i < n; i += stride) {
        int h   = i / (D * K);
        int rem = i - h * D * K;
        int d   = rem / K;
        int k   = rem - d * K;
        Wt[(size_t)h * D * K + (size_t)k * D + d] = (__bf16)W[i];
    }
}

// ---------------------------------------------------------------------------
// Kernel 3: per-head projection GEMM  Out[b,h] = X[b] (T x DM) * Wt[h]^T
// One wave per block computes a 32x64 tile (two M-tiles share B fragments).
// Register double-buffered K-loop over DM. transposeOut=1 -> [b,h][64][T].
// ---------------------------------------------------------------------------
__global__ void proj_kernel(const __bf16* __restrict__ X, const __bf16* __restrict__ Wt,
                            __bf16* __restrict__ Out, int transposeOut) {
    const int lane = threadIdx.x;
    const int half = lane >> 4;
    const int n    = lane & 15;
    const int tile = blockIdx.x;              // 32-row tile of T
    const int bh   = blockIdx.y;              // b*HHN + h
    const int b    = bh / HHN;
    const int h    = bh - b * HHN;

    const __bf16* arow0 = X + ((size_t)b * TT + (size_t)tile * 32 + n) * DMM;
    const __bf16* arow1 = arow0 + (size_t)16 * DMM;
    const __bf16* wcol  = Wt + ((size_t)h * 64 + n) * DMM;   // + t*16*DMM per N-tile

    v8f acc[2][4];
#pragma unroll
    for (int mt = 0; mt < 2; ++mt)
#pragma unroll
        for (int t = 0; t < 4; ++t) acc[mt][t] = (v8f){};

    v16bf a0 = load_a_frag(arow0, half, 0);
    v16bf a1 = load_a_frag(arow1, half, 0);
    v16bf bf[4];
#pragma unroll
    for (int t = 0; t < 4; ++t) bf[t] = load_b_frag(wcol + (size_t)t * 16 * DMM, half, 0);

    for (int k0 = 0; k0 < DMM; k0 += 32) {
        const int kn = (k0 + 32) & (DMM - 1);   // wraps to 0 on last iter (dead prefetch)
        v16bf na0 = load_a_frag(arow0, half, kn);
        v16bf na1 = load_a_frag(arow1, half, kn);
        v16bf nb[4];
#pragma unroll
        for (int t = 0; t < 4; ++t) nb[t] = load_b_frag(wcol + (size_t)t * 16 * DMM, half, kn);

#pragma unroll
        for (int t = 0; t < 4; ++t) {
            acc[0][t] = wmma_bf16(a0, bf[t], acc[0][t]);
            acc[1][t] = wmma_bf16(a1, bf[t], acc[1][t]);
        }
        a0 = na0; a1 = na1;
#pragma unroll
        for (int t = 0; t < 4; ++t) bf[t] = nb[t];
    }

    // C/D layout: element r -> row M = r + 8*half, col N = n
#pragma unroll
    for (int mt = 0; mt < 2; ++mt)
#pragma unroll
        for (int t = 0; t < 4; ++t)
#pragma unroll
            for (int r = 0; r < 8; ++r) {
                int row = tile * 32 + mt * 16 + r + 8 * half;
                int col = t * 16 + n;
                __bf16 v = (__bf16)acc[mt][t][r];
                if (!transposeOut)
                    Out[((size_t)bh * TT + row) * 64 + col] = v;
                else
                    Out[((size_t)bh * 64 + col) * TT + row] = v;
            }
}

// ---------------------------------------------------------------------------
// Kernel 4: flash attention, one wave per (b, h, 16-row q-tile).
// K/V chunks staged into LDS with async-to-LDS copies, double-buffered.
// ---------------------------------------------------------------------------
__global__ void attn_kernel(const __bf16* __restrict__ Q, const __bf16* __restrict__ Km,
                            const __bf16* __restrict__ Vt,
                            const float* __restrict__ vmask, const float* __restrict__ hmask,
                            __bf16* __restrict__ O) {
    __shared__ __bf16 kbuf[2][32 * 64];   // 32 keys x 64 dk
    __shared__ __bf16 vbuf[2][64 * 32];   // 64 dv x 32 keys
    __shared__ __bf16 pbuf[16 * 32];      // P tile, row-major

    const int lane  = threadIdx.x;
    const int half  = lane >> 4;
    const int n     = lane & 15;
    const int qtile = blockIdx.x;
    const int h     = blockIdx.y;
    const int b     = blockIdx.z;

    const __bf16* Qb  = Q  + (((size_t)b * HHN + h) * TT + (size_t)qtile * 16) * DKK;
    const __bf16* Kb  = Km + ((size_t)b * HHN + h) * TT * DKK;
    const __bf16* Vtb = Vt + ((size_t)b * HHN + h) * (size_t)DVV * TT;

    v16bf aq0 = load_a_frag(Qb + (size_t)n * DKK, half, 0);
    v16bf aq1 = load_a_frag(Qb + (size_t)n * DKK, half, 32);

    float hrow[8];
#pragma unroll
    for (int r = 0; r < 8; ++r)
        hrow[r] = hmask[(size_t)b * TT + qtile * 16 + r + 8 * half];

    v8f   o0 = {}, o1 = {}, o2 = {}, o3 = {};
    float mrow[8], lrow[8];
#pragma unroll
    for (int r = 0; r < 8; ++r) { mrow[r] = -__builtin_inff(); lrow[r] = 0.0f; }

    const float inv_dk = 1.0f / 64.0f;
    const int jcmax = (qtile * 16 + 15) / 32;

    stage_chunk(Kb, Vtb, 0, kbuf[0], vbuf[0], lane);

    for (int jc = 0; jc <= jcmax; ++jc) {
        const int base = jc * 32;
        const int cur  = jc & 1;
        if (jc < jcmax) {
            stage_chunk(Kb, Vtb, base + 32, kbuf[cur ^ 1], vbuf[cur ^ 1], lane);
            wait_async_16();   // current chunk's 16 copies complete (in-order)
        } else {
            wait_async_0();
        }
        const __bf16* kb = kbuf[cur];
        const __bf16* vb = vbuf[cur];

        // ---- S = Q * K^T for 32 keys (two 16-wide N tiles), K from LDS ----
        v8f s0 = {}, s1 = {};
        {
            const __bf16* kr0 = kb + (size_t)(0 * 16 + n) * 64;
            const __bf16* kr1 = kb + (size_t)(1 * 16 + n) * 64;
            s0 = wmma_bf16(aq0, load_b_frag(kr0, half, 0),  s0);
            s0 = wmma_bf16(aq1, load_b_frag(kr0, half, 32), s0);
            s1 = wmma_bf16(aq0, load_b_frag(kr1, half, 0),  s1);
            s1 = wmma_bf16(aq1, load_b_frag(kr1, half, 32), s1);
        }

        // ---- reference mask semantics + streaming softmax ----
        float sv[2][8];
        float vm0 = vmask[(size_t)b * TT + base + n];
        float vm1 = vmask[(size_t)b * TT + base + 16 + n];
#pragma unroll
        for (int r = 0; r < 8; ++r) {
            int qrow = qtile * 16 + r + 8 * half;
            float v0 = s0[r] * inv_dk * hrow[r] * vm0;
            if (v0 == 0.0f) v0 = 1e-10f;
            sv[0][r] = ((base + n) > qrow) ? -__builtin_inff() : v0;
            float v1 = s1[r] * inv_dk * hrow[r] * vm1;
            if (v1 == 0.0f) v1 = 1e-10f;
            sv[1][r] = ((base + 16 + n) > qrow) ? -__builtin_inff() : v1;
        }

        float scale[8], p0[8], p1[8];
#pragma unroll
        for (int r = 0; r < 8; ++r) {
            float rm = fmaxf(sv[0][r], sv[1][r]);
            for (int off = 8; off; off >>= 1)
                rm = fmaxf(rm, __shfl_xor(rm, off, 32));
            float mnew = fmaxf(mrow[r], rm);
            scale[r] = expf(mrow[r] - mnew);   // exp(-inf)=0 on first chunk
            p0[r] = expf(sv[0][r] - mnew);
            p1[r] = expf(sv[1][r] - mnew);
            lrow[r] = lrow[r] * scale[r] + p0[r] + p1[r];
            mrow[r] = mnew;
        }
#pragma unroll
        for (int r = 0; r < 8; ++r) {
            o0[r] *= scale[r]; o1[r] *= scale[r];
            o2[r] *= scale[r]; o3[r] *= scale[r];
        }

        // ---- P: C-layout -> LDS -> A-fragment ----
#pragma unroll
        for (int r = 0; r < 8; ++r) {
            pbuf[(r + 8 * half) * 32 + 0 * 16 + n] = (__bf16)p0[r];
            pbuf[(r + 8 * half) * 32 + 1 * 16 + n] = (__bf16)p1[r];
        }
        __syncthreads();
        v16bf pa = load_a_frag(&pbuf[(size_t)n * 32], half, 0);

        // ---- O += P * V, V from LDS (vb: [dv][32 keys]) ----
        o0 = wmma_bf16(pa, load_b_frag(vb + (size_t)(0 * 16 + n) * 32, half, 0), o0);
        o1 = wmma_bf16(pa, load_b_frag(vb + (size_t)(1 * 16 + n) * 32, half, 0), o1);
        o2 = wmma_bf16(pa, load_b_frag(vb + (size_t)(2 * 16 + n) * 32, half, 0), o2);
        o3 = wmma_bf16(pa, load_b_frag(vb + (size_t)(3 * 16 + n) * 32, half, 0), o3);
        __syncthreads();
    }

    // ---- finalize: divide by row sums, store O[b][t][h*64 + dv] ----
    v8f oo[4] = {o0, o1, o2, o3};
#pragma unroll
    for (int r = 0; r < 8; ++r) {
        float lr = lrow[r];
        for (int off = 8; off; off >>= 1)
            lr += __shfl_xor(lr, off, 32);
        float inv = 1.0f / lr;
        int row = qtile * 16 + r + 8 * half;
#pragma unroll
        for (int t = 0; t < 4; ++t)
            O[((size_t)b * TT + row) * (HHN * DVV) + h * DVV + t * 16 + n] =
                (__bf16)(oo[t][r] * inv);
    }
}

// ---------------------------------------------------------------------------
// Kernel 5: output projection  out[row][j] = sum_c Obuf[row][c] * w_out[j][c]
// 32x64 tile per wave, register double-buffered.
// ---------------------------------------------------------------------------
__global__ void outproj_kernel(const __bf16* __restrict__ Obuf, const __bf16* __restrict__ Wout,
                               float* __restrict__ out) {
    const int lane = threadIdx.x;
    const int half = lane >> 4;
    const int n    = lane & 15;
    const int tile = blockIdx.x;     // 32-row tile of B*T
    const int g    = blockIdx.y;     // 64-col group of DM

    const __bf16* arow0 = Obuf + ((size_t)tile * 32 + n) * (HHN * DVV);
    const __bf16* arow1 = arow0 + (size_t)16 * (HHN * DVV);
    const __bf16* wcol  = Wout + ((size_t)g * 64 + n) * DMM;

    v8f acc[2][4];
#pragma unroll
    for (int mt = 0; mt < 2; ++mt)
#pragma unroll
        for (int t = 0; t < 4; ++t) acc[mt][t] = (v8f){};

    v16bf a0 = load_a_frag(arow0, half, 0);
    v16bf a1 = load_a_frag(arow1, half, 0);
    v16bf bf[4];
#pragma unroll
    for (int t = 0; t < 4; ++t) bf[t] = load_b_frag(wcol + (size_t)t * 16 * DMM, half, 0);

    const int KTOT = HHN * DVV;      // 1024
    for (int k0 = 0; k0 < KTOT; k0 += 32) {
        const int kn = (k0 + 32) & (KTOT - 1);
        v16bf na0 = load_a_frag(arow0, half, kn);
        v16bf na1 = load_a_frag(arow1, half, kn);
        v16bf nb[4];
#pragma unroll
        for (int t = 0; t < 4; ++t) nb[t] = load_b_frag(wcol + (size_t)t * 16 * DMM, half, kn);

#pragma unroll
        for (int t = 0; t < 4; ++t) {
            acc[0][t] = wmma_bf16(a0, bf[t], acc[0][t]);
            acc[1][t] = wmma_bf16(a1, bf[t], acc[1][t]);
        }
        a0 = na0; a1 = na1;
#pragma unroll
        for (int t = 0; t < 4; ++t) bf[t] = nb[t];
    }

#pragma unroll
    for (int mt = 0; mt < 2; ++mt)
#pragma unroll
        for (int t = 0; t < 4; ++t)
#pragma unroll
            for (int r = 0; r < 8; ++r) {
                int row = tile * 32 + mt * 16 + r + 8 * half;
                out[(size_t)row * DMM + g * 64 + t * 16 + n] = acc[mt][t][r];
            }
}

// ---------------------------------------------------------------------------
// Host launcher
// ---------------------------------------------------------------------------
extern "C" void kernel_launch(void* const* d_in, const int* in_sizes, int n_in,
                              void* d_out, int out_size, void* d_ws, size_t ws_size,
                              hipStream_t stream) {
    const float* xq    = (const float*)d_in[0];
    const float* xk    = (const float*)d_in[1];
    const float* xv    = (const float*)d_in[2];
    const float* vmask = (const float*)d_in[3];
    const float* hmask = (const float*)d_in[4];
    const float* w_q   = (const float*)d_in[5];
    const float* w_k   = (const float*)d_in[6];
    const float* w_v   = (const float*)d_in[7];
    const float* w_out = (const float*)d_in[8];
    float* out = (float*)d_out;

    char* ws = (char*)d_ws;
    size_t off = 0;
    auto alloc = [&](size_t bytes) -> void* {
        void* p = ws + off;
        off += (bytes + 255) & ~(size_t)255;
        return p;
    };

    const size_t nX  = (size_t)BB * TT * DMM;          // 4M
    const size_t nW  = (size_t)HHN * DMM * DKK;        // 1M
    const size_t nWo = (size_t)DMM * HHN * DVV;        // 1M
    const size_t nQ  = (size_t)BB * HHN * TT * DKK;    // 4M

    __bf16* Xq  = (__bf16*)alloc(nX * 2);
    __bf16* Xk  = (__bf16*)alloc(nX * 2);
    __bf16* Xv  = (__bf16*)alloc(nX * 2);
    __bf16* Wqt = (__bf16*)alloc(nW * 2);
    __bf16* Wkt = (__bf16*)alloc(nW * 2);
    __bf16* Wvt = (__bf16*)alloc(nW * 2);
    __bf16* Wob = (__bf16*)alloc(nWo * 2);
    __bf16* Qb  = (__bf16*)alloc(nQ * 2);
    __bf16* Kb  = (__bf16*)alloc(nQ * 2);
    __bf16* Vtb = (__bf16*)alloc(nQ * 2);
    __bf16* Ob  = (__bf16*)alloc(nX * 2);

    // 1) bf16 conversions
    {
        int blk = 256;
        cvt_bf16_kernel<<<2048, blk, 0, stream>>>(xq,    Xq,  (int)nX);
        cvt_bf16_kernel<<<2048, blk, 0, stream>>>(xk,    Xk,  (int)nX);
        cvt_bf16_kernel<<<2048, blk, 0, stream>>>(xv,    Xv,  (int)nX);
        cvt_bf16_kernel<<<2048, blk, 0, stream>>>(w_out, Wob, (int)nWo);
        cvt_transpose_w_kernel<<<2048, blk, 0, stream>>>(w_q, Wqt, DMM, DKK, (int)nW);
        cvt_transpose_w_kernel<<<2048, blk, 0, stream>>>(w_k, Wkt, DMM, DKK, (int)nW);
        cvt_transpose_w_kernel<<<2048, blk, 0, stream>>>(w_v, Wvt, DMM, DVV, (int)nW);
    }

    // 2) Q/K/V projections (V transposed to [dv][T])
    {
        dim3 grid(TT / 32, BB * HHN);
        proj_kernel<<<grid, 32, 0, stream>>>(Xq, Wqt, Qb,  0);
        proj_kernel<<<grid, 32, 0, stream>>>(Xk, Wkt, Kb,  0);
        proj_kernel<<<grid, 32, 0, stream>>>(Xv, Wvt, Vtb, 1);
    }

    // 3) flash attention
    {
        dim3 grid(TT / 16, HHN, BB);
        attn_kernel<<<grid, 32, 0, stream>>>(Qb, Kb, Vtb, vmask, hmask, Ob);
    }

    // 4) output projection -> f32 d_out
    {
        dim3 grid((BB * TT) / 32, DMM / 64);
        outproj_kernel<<<grid, 32, 0, stream>>>(Ob, Wob, out);
    }

    (void)in_sizes; (void)n_in; (void)out_size; (void)ws_size;
}
